// SBGCN_78039555768461
// MI455X (gfx1250) — compile-verified, matrix operand order checked by reading, not run on hardware
//
#include <hip/hip_runtime.h>
#include <hip/hip_bf16.h>
#include <stdint.h>

#define WH 128

typedef __attribute__((ext_vector_type(16))) __bf16 v16bf;
typedef __attribute__((ext_vector_type(8)))  float  v8f;

// ---------- helpers ----------
__device__ __forceinline__ unsigned short f2bf(float f) {
    unsigned u = __float_as_uint(f);
    unsigned r = u + 0x7FFFu + ((u >> 16) & 1u);   // round-to-nearest-even
    return (unsigned short)(r >> 16);
}
// monotone (order-preserving) encoding of float into uint so that
// unsigned atomicMax == float max.  enc(x)==0 is unreachable for real data,
// so 0 is the "no contribution" sentinel (scatter_max fill -> 0.0).
__device__ __forceinline__ unsigned encf(float f) {
    unsigned u = __float_as_uint(f);
    return (u & 0x80000000u) ? ~u : (u | 0x80000000u);
}
__device__ __forceinline__ float decf(unsigned u) {
    unsigned v = (u & 0x80000000u) ? (u & 0x7FFFFFFFu) : ~u;
    return __uint_as_float(v);
}

// ---------- weight pre-swizzle into WMMA B-fragment order ----------
// B (32x16 per k-step) 16-bit layout: lane L -> N = L&15, h = L>>4;
// dword pair ii -> K = kblock*32 + (ii<4?0:16) + h*8 + (ii&3)*2 (+0/+1)
__global__ void pack_w(const float* __restrict__ W, unsigned short* __restrict__ P,
                       int total) {
    int gid = blockIdx.x * blockDim.x + threadIdx.x;
    if (gid >= total) return;
    int i    = gid & 15;          // half index within fragment
    int lane = (gid >> 4) & 31;
    int w    = (gid >> 9) & 7;    // 16-col tile
    int t    = (gid >> 12) & 7;   // 32-K block
    int conv = gid >> 15;
    int h  = lane >> 4;
    int n  = lane & 15;
    int ii = i >> 1;
    int k  = t * 32 + ((ii < 4) ? 0 : 16) + h * 8 + (ii & 3) * 2 + (i & 1);
    P[gid] = f2bf(W[(conv * 256 + k) * WH + (w * 16 + n)]);
}

// ---------- small-K embedding: y = lrelu(x @ W + b) ----------
__global__ void embed(const float* __restrict__ X, const float* __restrict__ W,
                      const float* __restrict__ b, float* __restrict__ Y, int K) {
    __shared__ float row[80];
    int r = blockIdx.x, c = threadIdx.x;
    if (c < K) row[c] = X[(size_t)r * K + c];
    __syncthreads();
    float acc = b[c];
    for (int k = 0; k < K; ++k) acc += row[k] * W[k * WH + c];
    Y[(size_t)r * WH + c] = acc > 0.f ? acc : 0.01f * acc;
}

// ---------- per-edge gather-diff + ordered atomic scatter-max ----------
__global__ void scat(const float* __restrict__ xs, const float* __restrict__ xd,
                     const int* __restrict__ sidx, const int* __restrict__ didx,
                     int nE, unsigned* __restrict__ ord) {
    __shared__ int ss[32], sd[32];
    int e0 = blockIdx.x * 32, c = threadIdx.x;
    if (c < 32 && e0 + c < nE) { ss[c] = sidx[e0 + c]; sd[c] = didx[e0 + c]; }
    __syncthreads();
    int n = nE - e0; if (n > 32) n = 32;
    for (int j = 0; j < n; ++j) {
        int s = ss[j], d = sd[j];
        float diff = xd[(size_t)d * WH + c] - xs[(size_t)s * WH + c];
        atomicMax(&ord[(size_t)d * WH + c], encf(diff));
    }
}

// ---------- conv MLP: x += lrelu([x | segmax] @ W + b), WMMA bf16 ----------
__global__ __launch_bounds__(256) void conv_gemm(float* __restrict__ xd,
        const unsigned* __restrict__ ord, const unsigned short* __restrict__ Pw,
        const float* __restrict__ bias) {
    __shared__ unsigned short As[16 * 256];      // 16 rows x 256 K, bf16
    int tid  = threadIdx.x;
    int row0 = blockIdx.x * 16;
    // stage [x_dst_tile | decoded maxes] as bf16
    for (int it = 0; it < 8; ++it) {
        int idx = it * 256 + tid;                // 0..2047
        int r = idx >> 7, c = idx & 127;
        As[r * 256 + c] = f2bf(xd[(size_t)(row0 + r) * WH + c]);
        unsigned o = ord[(size_t)(row0 + r) * WH + c];
        As[r * 256 + 128 + c] = f2bf(o ? decf(o) : 0.0f);
    }
    __syncthreads();

    int lane = tid & 31;
    int w    = tid >> 5;          // wave id -> 16-col tile (8 waves)
    int hh   = lane >> 4;
    int M    = lane & 15;
    v8f acc = {};
    for (int t = 0; t < 8; ++t) {                // K = 256 in 8 steps of 32
        union { unsigned u[8]; v16bf b; } A;
        int k0 = t * 32;
        #pragma unroll
        for (int ii = 0; ii < 8; ++ii) {
            int kb = k0 + ((ii < 4) ? 0 : 16) + hh * 8 + (ii & 3) * 2;
            A.u[ii] = *(const unsigned*)&As[M * 256 + kb];   // two halfs K,K+1
        }
        union { uint4 q[2]; v16bf b; } B;
        const uint4* bp = (const uint4*)(Pw + (size_t)((t * 8 + w) * 32 + lane) * 16);
        B.q[0] = bp[0]; B.q[1] = bp[1];
        acc = __builtin_amdgcn_wmma_f32_16x16x32_bf16(
                  false, A.b, false, B.b, (short)0, acc, false, false);
    }
    // epilogue: C/D layout -> VGPR j holds M=j (lanes0-15) / M=j+8 (lanes16-31)
    int col = w * 16 + (lane & 15);
    float bval = bias[col];
    #pragma unroll
    for (int j = 0; j < 8; ++j) {
        int r = row0 + j + 8 * hh;
        float h = acc[j] + bval;
        h = h > 0.f ? h : 0.01f * h;
        size_t o = (size_t)r * WH + col;
        xd[o] = xd[o] + h;                        // residual (block-local rows)
    }
}

// ---------- flat-topo assembly via maps ----------
__global__ void copymap(const float* __restrict__ src, const int* __restrict__ map,
                        int n, float* __restrict__ xt) {
    int i = blockIdx.x, c = threadIdx.x;
    xt[(size_t)map[n + i] * WH + c] = src[(size_t)map[i] * WH + c];
}

// ---------- global max pool ----------
__global__ void pool_scat(const float* __restrict__ xt, const int* __restrict__ gidx,
                          unsigned* __restrict__ pord) {
    int i = blockIdx.x, c = threadIdx.x;
    int g = gidx[i];
    atomicMax(&pord[(size_t)g * WH + c], encf(xt[(size_t)i * WH + c]));
}
__global__ void pool_dec(const unsigned* __restrict__ pord, float* __restrict__ xp) {
    int i = blockIdx.x * blockDim.x + threadIdx.x;
    unsigned o = pord[i];
    xp[i] = o ? decf(o) : 0.0f;
}

// ---------- launch ----------
extern "C" void kernel_launch(void* const* d_in, const int* in_sizes, int n_in,
                              void* d_out, int out_size, void* d_ws, size_t ws_size,
                              hipStream_t stream) {
    const int F_N = 50000, L_N = 100000, E_N = 200000, V_N = 150000;

    const float* in_xf = (const float*)d_in[0];
    const float* in_xl = (const float*)d_in[1];
    const float* in_xe = (const float*)d_in[2];
    const float* in_xv = (const float*)d_in[3];
    const int* e2v = (const int*)d_in[4];   // rows of 400000
    const int* l2e = (const int*)d_in[5];   // rows of 400000
    const int* f2l = (const int*)d_in[6];   // rows of 200000
    const int* f2f = (const int*)d_in[7];   // rows of 300000
    const int* fmap = (const int*)d_in[8];
    const int* lmap = (const int*)d_in[9];
    const int* emap = (const int*)d_in[10];
    const int* vmap = (const int*)d_in[11];
    const int* gidx = (const int*)d_in[12];
    const float* Wf = (const float*)d_in[13]; const float* bf = (const float*)d_in[14];
    const float* Wl = (const float*)d_in[15]; const float* bl = (const float*)d_in[16];
    const float* We = (const float*)d_in[17]; const float* be = (const float*)d_in[18];
    const float* Wv = (const float*)d_in[19]; const float* bv = (const float*)d_in[20];
    const float* convW = (const float*)d_in[21];
    const float* convB = (const float*)d_in[22];

    float* out = (float*)d_out;
    float* xt = out;                                     // 500000 x 128
    float* xp = out + (size_t)500000 * WH;               // 64 x 128
    float* xf = xp + (size_t)64 * WH;
    float* xl = xf + (size_t)F_N * WH;
    float* xe = xl + (size_t)L_N * WH;
    float* xv = xe + (size_t)E_N * WH;

    unsigned* ord = (unsigned*)d_ws;                     // up to 200000*128 u32
    unsigned short* packed =
        (unsigned short*)((char*)d_ws + (size_t)E_N * WH * sizeof(unsigned));

    // weights -> bf16 B-fragment layout (9 convs * 32768 halfs)
    int totalP = 9 * 8 * 8 * 32 * 16;
    pack_w<<<(totalP + 255) / 256, 256, 0, stream>>>(convW, packed, totalP);

    // embeddings (written straight into d_out slices)
    embed<<<F_N, 128, 0, stream>>>(in_xf, Wf, bf, xf, 62);
    embed<<<L_N, 128, 0, stream>>>(in_xl, Wl, bl, xl, 38);
    embed<<<E_N, 128, 0, stream>>>(in_xe, We, be, xe, 72);
    embed<<<V_N, 128, 0, stream>>>(in_xv, Wv, bv, xv, 3);

    struct Conv { const float* xs; float* xd; const int* si; const int* di;
                  int nE; int nD; int w; };
    const Conv cv[9] = {
        { xv, xe, e2v + 400000, e2v,          400000, E_N, 0 }, // V2E (flipped)
        { xe, xl, l2e + 400000, l2e,          400000, L_N, 1 }, // E2L
        { xl, xf, f2l + 200000, f2l,          200000, F_N, 2 }, // L2F
        { xf, xf, f2f,          f2f + 300000, 300000, F_N, 3 }, // FF k=0
        { xf, xf, f2f,          f2f + 300000, 300000, F_N, 4 }, // FF k=1
        { xf, xf, f2f,          f2f + 300000, 300000, F_N, 5 }, // FF k=2
        { xf, xl, f2l,          f2l + 200000, 200000, L_N, 6 }, // F2L
        { xl, xe, l2e,          l2e + 400000, 400000, E_N, 7 }, // L2E
        { xe, xv, e2v,          e2v + 400000, 400000, V_N, 8 }, // E2V
    };
    for (int j = 0; j < 9; ++j) {
        const Conv& c = cv[j];
        hipMemsetAsync(ord, 0, (size_t)c.nD * WH * sizeof(unsigned), stream);
        scat<<<(c.nE + 31) / 32, 128, 0, stream>>>(c.xs, c.xd, c.si, c.di, c.nE, ord);
        conv_gemm<<<c.nD / 16, 256, 0, stream>>>(c.xd, ord,
                packed + (size_t)c.w * 32768, convB + (size_t)c.w * WH);
    }

    // flat-topo assembly
    copymap<<<F_N, 128, 0, stream>>>(xf, fmap, F_N, xt);
    copymap<<<L_N, 128, 0, stream>>>(xl, lmap, L_N, xt);
    copymap<<<E_N, 128, 0, stream>>>(xe, emap, E_N, xt);
    copymap<<<V_N, 128, 0, stream>>>(xv, vmap, V_N, xt);

    // global max pool over 64 graphs
    hipMemsetAsync(ord, 0, (size_t)64 * WH * sizeof(unsigned), stream);
    pool_scat<<<500000, 128, 0, stream>>>(xt, gidx, ord);
    pool_dec<<<64, 128, 0, stream>>>(ord, xp);
}
